// DiffractiveNetwork_18330920419445
// MI455X (gfx1250) — matrix operand order repaired and checked by reading, not compiled, not run on hardware
//
#include <hip/hip_runtime.h>
#include <cmath>

typedef __attribute__((ext_vector_type(16))) _Float16 v16h;
typedef __attribute__((ext_vector_type(8)))  _Float16 v8h;
typedef __attribute__((ext_vector_type(8)))  float    v8f;

#define NN    4096
#define BATCH 32
#define KSPLIT 8            // waves per workgroup, each owns NN/KSPLIT of K

#define PI_F       3.14159265358979323846f
#define LAMBDA0f   1.55e-6f
#define W0f        0.45e-6f
#define H_NEURONf  3.0e-6f
#define DELTAf     1.0e-7f
// LAMBDA/(PI*W0^2), with LAMBDA = LAMBDA0/2.85
#define GBf        (LAMBDA0f / 2.85f / (PI_F * W0f * W0f))
#define K_SUBf     (2.0f * PI_F * 1.444f / LAMBDA0f)

static __device__ __forceinline__ v8f wmma_f16(v16h a, v16h b, v8f c) {
    return __builtin_amdgcn_wmma_f32_16x16x32_f16(
        /*neg_a=*/false, a, /*neg_b=*/false, b,
        /*c_mod=*/(short)0, c, /*reuse_a=*/false, /*reuse_b=*/false);
}

// Row-major f16 A-tile fragment: per ISA A-layout, lane (l16, hseg) holds the
// two contiguous 8-half runs K = hseg*8+0..7 and K = 16+hseg*8+0..7 of row l16.
static __device__ __forceinline__ v16h load_a_frag(const _Float16* __restrict__ base,
                                                   int row, int k0, int hseg) {
    const _Float16* p = base + (size_t)row * NN + k0 + hseg * 8;
    v8h lo = *(const v8h*)(p);
    v8h hi = *(const v8h*)(p + 16);
    return __builtin_shufflevector(lo, hi, 0, 1, 2, 3, 4, 5, 6, 7,
                                   8, 9, 10, 11, 12, 13, 14, 15);
}

// ---------------------------------------------------------------------------
// init: Wr = waves (f32 -> f16), Wi = 0
// ---------------------------------------------------------------------------
__global__ void diffnet_init_kernel(const float* __restrict__ waves,
                                    _Float16* __restrict__ Wr,
                                    _Float16* __restrict__ Wi) {
    int i = blockIdx.x * 256 + threadIdx.x;
    if (i < BATCH * NN) {
        Wr[i] = (_Float16)waves[i];
        Wi[i] = (_Float16)0.0f;
    }
}

// ---------------------------------------------------------------------------
// Fused layer: W_out[b,m] = sum_l W_in[b,l] * C(m,l)   (complex)
// C(m,l) generated on the fly from coordinates.
// Block = 256 threads = 8 waves, all sharing ONE 16-neuron output tile.
// Wave w covers K in [w*512, (w+1)*512); partials reduced in LDS in a fixed
// (deterministic) order. Grid = 256 blocks -> 2048 waves: fills the chip.
// ---------------------------------------------------------------------------
__global__ void __launch_bounds__(256)
diffnet_layer_kernel(const _Float16* __restrict__ Wr_in,
                     const _Float16* __restrict__ Wi_in,
                     const float* __restrict__ x0,
                     const float* __restrict__ y0,
                     const float* __restrict__ xn,
                     const float* __restrict__ yn,
                     _Float16* __restrict__ Wr_out,
                     _Float16* __restrict__ Wi_out,
                     float prefR, float prefI, float cRsm) {
    __shared__ float spart[KSPLIT][4 * 8 * 32];   // 32 KB partial accumulators

    const int tid  = threadIdx.x;
    const int lane = tid & 31;
    const int wid  = tid >> 5;
    const int hseg = lane >> 4;      // 0: lanes 0-15, 1: lanes 16-31
    const int l16  = lane & 15;
    const int m    = blockIdx.x * 16 + l16;     // output neuron (N col)
    const int kbase = wid * (NN / KSPLIT);

    const float xnm = xn[m];
    const float ynm = yn[m];

    v8f accR0 = {}, accI0 = {}, accR1 = {}, accI1 = {};

    for (int chunk = 0; chunk < (NN / KSPLIT) / 32; ++chunk) {
        const int k0 = kbase + chunk * 32;

        // --- A fragments: raw 16B f16 loads in WMMA layout (no conversion)
        v16h a0r = load_a_frag(Wr_in, l16,      k0, hseg);
        v16h a0i = load_a_frag(Wi_in, l16,      k0, hseg);
        v16h a1r = load_a_frag(Wr_in, l16 + 16, k0, hseg);
        v16h a1i = load_a_frag(Wi_in, l16 + 16, k0, hseg);

        // --- source coordinates for this half-wave's 16 l-values (two 8-runs: hseg*8+0..7, 16+hseg*8+0..7)
        float xs[16], ys[16];
        {
            const float4* px = (const float4*)(x0 + k0 + hseg * 8);
            const float4* py = (const float4*)(y0 + k0 + hseg * 8);
#pragma unroll
            for (int q = 0; q < 4; ++q) {
                const int qq = (q < 2) ? q : q + 2;
                float4 vx = px[qq], vy = py[qq];
                xs[q * 4 + 0] = vx.x; xs[q * 4 + 1] = vx.y;
                xs[q * 4 + 2] = vx.z; xs[q * 4 + 3] = vx.w;
                ys[q * 4 + 0] = vy.x; ys[q * 4 + 1] = vy.y;
                ys[q * 4 + 2] = vy.z; ys[q * 4 + 3] = vy.w;
            }
        }

        // --- B fragments: generate coupling C(m, l), l = k0 + 8*hseg + e (e<8), k0 + 16 + 8*hseg + (e-8) (e>=8)
        v16h bcr, bci, bnci;
#pragma unroll
        for (int e = 0; e < 16; ++e) {
            const float r0 = xnm - xs[e];
            const float z  = fabsf(ynm - ys[e] + (H_NEURONf + DELTAf));
            const float r2 = fmaf(r0, r0, z * z);
            const float ir = rsqrtf(r2);          // 1/r
            const float r  = r2 * ir;             // r
            // e_rsm = cRsm * sqrt(1/r) * cos_theta,  cos_theta = z/r
            const float ersm = cRsm * z * ir * sqrtf(ir);
            // e_gbm = sqrt(W0/w) * exp(-r0^2/w^2),  w^2 = W0^2 (1 + (z*GB)^2)
            const float t   = z * GBf;
            const float w2  = (W0f * W0f) * fmaf(t, t, 1.0f);
            const float iw  = rsqrtf(w2);         // 1/w
            const float egbm = sqrtf(W0f * iw) * __expf(-(r0 * r0) * (iw * iw));
            const float amp  = ersm + egbm;
            // phase = exp(-i*K_SUB*r);  C = PREF * amp * (cos - i sin)
            float s, c;
            __sincosf(K_SUBf * r, &s, &c);
            const float cr = amp * fmaf(prefR, c, prefI * s);
            const float ci = amp * fmaf(prefI, c, -prefR * s);
            bcr[e]  = (_Float16)cr;
            bci[e]  = (_Float16)ci;
            bnci[e] = (_Float16)(-ci);
        }

        // --- complex GEMM via 8 real WMMAs (f16 in, f32 accumulate)
        accR0 = wmma_f16(a0r, bcr,  accR0);
        accR0 = wmma_f16(a0i, bnci, accR0);
        accI0 = wmma_f16(a0r, bci,  accI0);
        accI0 = wmma_f16(a0i, bcr,  accI0);
        accR1 = wmma_f16(a1r, bcr,  accR1);
        accR1 = wmma_f16(a1i, bnci, accR1);
        accI1 = wmma_f16(a1r, bci,  accI1);
        accI1 = wmma_f16(a1i, bcr,  accI1);
    }

    // --- deterministic cross-wave K reduction in LDS
#pragma unroll
    for (int g = 0; g < 8; ++g) {
        spart[wid][(0 * 8 + g) * 32 + lane] = accR0[g];
        spart[wid][(1 * 8 + g) * 32 + lane] = accI0[g];
        spart[wid][(2 * 8 + g) * 32 + lane] = accR1[g];
        spart[wid][(3 * 8 + g) * 32 + lane] = accI1[g];
    }
    __syncthreads();

    for (int i = tid; i < 4 * 8 * 32; i += 256) {
        float s = spart[0][i];
#pragma unroll
        for (int w = 1; w < KSPLIT; ++w) s += spart[w][i];
        // decode: i = ((acc*8)+g)*32 + ln ; acc: 0=R0 1=I0 2=R1 3=I1
        const int acc = i >> 8;
        const int g   = (i >> 5) & 7;
        const int ln  = i & 31;
        const int hs  = ln >> 4;
        // D layout: lanes 0-15 -> M=g, lanes 16-31 -> M=8+g; N = ln&15
        const int row = g + 8 * hs + ((acc >> 1) ? 16 : 0);
        const int col = blockIdx.x * 16 + (ln & 15);
        const _Float16 h = (_Float16)s;
        if (acc & 1) Wi_out[(size_t)row * NN + col] = h;
        else         Wr_out[(size_t)row * NN + col] = h;
    }
}

// ---------------------------------------------------------------------------
// softmax(|wc|) over each batch row (reads f16 wave planes)
// ---------------------------------------------------------------------------
__global__ void __launch_bounds__(256)
diffnet_softmax_kernel(const _Float16* __restrict__ Wr,
                       const _Float16* __restrict__ Wi,
                       float* __restrict__ out) {
    __shared__ float red[256];
    const int b = blockIdx.x;
    const int t = threadIdx.x;
    const _Float16* rr = Wr + (size_t)b * NN;
    const _Float16* ii = Wi + (size_t)b * NN;

    float mx = -3.4e38f;
    for (int i = t; i < NN; i += 256) {
        const float re = (float)rr[i], im = (float)ii[i];
        mx = fmaxf(mx, sqrtf(re * re + im * im));
    }
    red[t] = mx;
    __syncthreads();
    for (int s = 128; s > 0; s >>= 1) {
        if (t < s) red[t] = fmaxf(red[t], red[t + s]);
        __syncthreads();
    }
    mx = red[0];
    __syncthreads();

    float sum = 0.0f;
    for (int i = t; i < NN; i += 256) {
        const float re = (float)rr[i], im = (float)ii[i];
        sum += __expf(sqrtf(re * re + im * im) - mx);
    }
    red[t] = sum;
    __syncthreads();
    for (int s = 128; s > 0; s >>= 1) {
        if (t < s) red[t] += red[t + s];
        __syncthreads();
    }
    const float inv = 1.0f / red[0];

    for (int i = t; i < NN; i += 256) {
        const float re = (float)rr[i], im = (float)ii[i];
        out[(size_t)b * NN + i] = __expf(sqrtf(re * re + im * im) - mx) * inv;
    }
}

// ---------------------------------------------------------------------------
extern "C" void kernel_launch(void* const* d_in, const int* in_sizes, int n_in,
                              void* d_out, int out_size, void* d_ws, size_t ws_size,
                              hipStream_t stream) {
    (void)in_sizes; (void)n_in; (void)out_size; (void)ws_size;
    const float* waves = (const float*)d_in[0];
    const float* x0_0  = (const float*)d_in[1];
    const float* y0_0  = (const float*)d_in[2];
    const float* x0_1  = (const float*)d_in[3];
    const float* y0_1  = (const float*)d_in[4];
    const float* x0_2  = (const float*)d_in[5];
    const float* y0_2  = (const float*)d_in[6];
    const float* x_out = (const float*)d_in[7];
    const float* y_out = (const float*)d_in[8];
    float* out = (float*)d_out;

    // ping-pong complex f16 wave planes in workspace (4 x 256 KB)
    const size_t S = (size_t)BATCH * NN;
    _Float16* Ar = (_Float16*)d_ws;
    _Float16* Ai = Ar + S;
    _Float16* Br = Ar + 2 * S;
    _Float16* Bi = Ar + 3 * S;

    // scalar complex prefactor: F * exp(-i*beta*H/2) * eta * exp(i*phi)
    const double beta = 2.0 * M_PI * 2.85 / 1.55e-6;
    const double ang  = -beta * 3.0e-6 / 2.0;
    const float prefR = (float)std::cos(ang);
    const float prefI = (float)std::sin(ang);
    // K_RSM * sqrt(2*W0/sqrt(pi))
    const float cRsm  = (float)std::sqrt(2.0 * 0.45e-6 / std::sqrt(M_PI));

    diffnet_init_kernel<<<(int)((S + 255) / 256), 256, 0, stream>>>(waves, Ar, Ai);

    dim3 lgrid(NN / 16), lblk(256);
    diffnet_layer_kernel<<<lgrid, lblk, 0, stream>>>(Ar, Ai, x0_0, y0_0, x0_1, y0_1,
                                                     Br, Bi, prefR, prefI, cRsm);
    diffnet_layer_kernel<<<lgrid, lblk, 0, stream>>>(Br, Bi, x0_1, y0_1, x0_2, y0_2,
                                                     Ar, Ai, prefR, prefI, cRsm);
    diffnet_layer_kernel<<<lgrid, lblk, 0, stream>>>(Ar, Ai, x0_2, y0_2, x_out, y_out,
                                                     Br, Bi, prefR, prefI, cRsm);

    diffnet_softmax_kernel<<<BATCH, 256, 0, stream>>>(Br, Bi, out);
}
